// TransformerEncoderLayer_31129922961436
// MI455X (gfx1250) — matrix operand order repaired and checked by reading, not compiled, not run on hardware
//
#include <hip/hip_runtime.h>
#include <hip/hip_bf16.h>

// ---------------------------------------------------------------------------
// Transformer encoder layer for MI455X (gfx1250), wave32 + WMMA bf16 +
// async-to-LDS staging (ASYNCcnt) + TDM tensor_load (TENSORcnt).
// B=2 S=2048 D=1024 H=16 HD=64 FF=4096.
// ---------------------------------------------------------------------------

#define B_  2
#define S_  2048
#define D_  1024
#define H_  16
#define HD_ 64
#define FF_ 4096
#define MROWS (B_ * S_)   // 4096

typedef __attribute__((ext_vector_type(16))) __bf16 bf16x16;
typedef __attribute__((ext_vector_type(8)))  float  floatx8;
typedef __attribute__((ext_vector_type(4)))  unsigned int uint32x4;
typedef __attribute__((ext_vector_type(8)))  int  int32x8;
typedef __attribute__((ext_vector_type(4)))  int  int32x4;

__device__ __forceinline__ floatx8 wmma_bf16(bf16x16 a, bf16x16 b, floatx8 c) {
  return __builtin_amdgcn_wmma_f32_16x16x32_bf16(false, a, false, b, (short)0, c,
                                                 false, false);
}

__device__ __forceinline__ unsigned short f2bf(float f) {
  union { float f; unsigned u; } v; v.f = f;
  unsigned r = (v.u + 0x7FFFu + ((v.u >> 16) & 1u)) >> 16;   // RNE
  return (unsigned short)r;
}
__device__ __forceinline__ float bf2f(unsigned short h) {
  union { unsigned u; float f; } v; v.u = ((unsigned)h) << 16;
  return v.f;
}

// Wave-relative LDS byte address (low 32 bits of the generic pointer).
__device__ __forceinline__ unsigned lds_addr_of(const void* p) {
  return (unsigned)(unsigned long long)p;
}

// 16B global -> 16B LDS, per-lane addresses, tracked by ASYNCcnt.
__device__ __forceinline__ void async_b128(unsigned lds_byte, const void* gptr) {
  asm volatile("global_load_async_to_lds_b128 %0, %1, off"
               :: "v"(lds_byte), "v"(gptr) : "memory");
}
__device__ __forceinline__ void async_wait() {
  asm volatile("s_wait_asynccnt 0x0" ::: "memory");
}

// Fragment-order maps used here (16-bit A 16x32 / B 32x16; contiguous k order for both):
//  A element (m,k): lane=(m&15)+16*(k>=16), slot=k&15
//   -> lane chunk c (slots 8c..8c+7) covers K = 16*(lane>>4) + 8c + [0..7]
//  B element (k,n): lane=(n&15)+16*(k>=16), slot=k&15
//   -> lane chunk c covers K = 16*(lane>>4) + 8c + [0..7]
// Both chunk K-ranges are contiguous in row-major global memory, so the
// fragment-order LDS image is built directly with async b128 copies.

// ---------------------------------------------------------------------------
// f32 -> bf16 conversion
// ---------------------------------------------------------------------------
__global__ __launch_bounds__(256) void cvt_kernel(const float* __restrict__ src,
                                                  unsigned short* __restrict__ dst,
                                                  int n) {
  int i = blockIdx.x * 256 + threadIdx.x;
  if (i < n) dst[i] = f2bf(src[i]);
}

// ---------------------------------------------------------------------------
// LayerNorm: one row (D=1024) per 256-thread block, bf16 output.
// ---------------------------------------------------------------------------
__global__ __launch_bounds__(256) void ln_kernel(const float* __restrict__ x,
                                                 const float* __restrict__ w,
                                                 const float* __restrict__ b,
                                                 unsigned short* __restrict__ out) {
  __shared__ float rs[8], rq[8], mv[2];
  const int tid = threadIdx.x;
  const size_t row = blockIdx.x;
  const float* xr = x + row * D_;
  float vals[4], s = 0.f, q = 0.f;
  for (int i = 0; i < 4; i++) {
    float v = xr[tid + i * 256];
    vals[i] = v; s += v; q += v * v;
  }
  for (int m = 16; m >= 1; m >>= 1) { s += __shfl_xor(s, m, 32); q += __shfl_xor(q, m, 32); }
  if ((tid & 31) == 0) { rs[tid >> 5] = s; rq[tid >> 5] = q; }
  __syncthreads();
  if (tid == 0) {
    float S = 0.f, Q = 0.f;
    for (int i = 0; i < 8; i++) { S += rs[i]; Q += rq[i]; }
    float mean = S * (1.0f / D_);
    mv[0] = mean;
    mv[1] = rsqrtf(Q * (1.0f / D_) - mean * mean + 1e-5f);
  }
  __syncthreads();
  float mean = mv[0], inv = mv[1];
  for (int i = 0; i < 4; i++) {
    int c = tid + i * 256;
    out[row * D_ + c] = f2bf((vals[i] - mean) * inv * w[c] + b[c]);
  }
}

// ---------------------------------------------------------------------------
// WMMA GEMM: C[M,N] = A[M,K](bf16)*B[K,N](bf16) + bias (+GELU/+residual),
// 128x128 tile, 8 waves, BK=32.  A tile staged with async b128 directly in
// fragment order; B tile scatter-staged; software-pipelined k-loop.
// ---------------------------------------------------------------------------
template <int GELU, int RES, int OUTBF>
__global__ __launch_bounds__(256) void gemm_kernel(
    const unsigned short* __restrict__ A, const unsigned short* __restrict__ Bm,
    const float* __restrict__ bias, const float* __restrict__ res,
    float* __restrict__ outF, unsigned short* __restrict__ outB,
    int M, int N, int K) {
  __shared__ __attribute__((aligned(16))) unsigned short lA[8 * 512];
  __shared__ __attribute__((aligned(16))) unsigned short lB[8 * 512];
  const int tid = threadIdx.x;
  const int lane = tid & 31, wave = tid >> 5;
  const int wm = wave >> 1, wn = wave & 1;         // 4x2 wave grid
  const int rowBase = blockIdx.y * 128;
  const int colBase = blockIdx.x * 128;

  floatx8 acc[2][4];
  for (int i = 0; i < 2; i++)
    for (int j = 0; j < 4; j++) acc[i][j] = {};

  auto stageA = [&](int kk) {   // 512 x 16B chunks, async, fragment order
    for (int e = tid; e < 512; e += 256) {
      int subtile = e >> 6, ls = (e >> 1) & 31, c = e & 1;
      int m = subtile * 16 + (ls & 15);
      int kf = 8 * c + 16 * (ls >> 4);
      async_b128(lds_addr_of(&lA[subtile * 512 + ls * 16 + c * 8]),
                 &A[(size_t)(rowBase + m) * K + kk + kf]);
    }
  };
  auto stageB = [&](int kk) {   // scatter (column-of-K per lane is strided)
    for (int e = tid; e < 4096; e += 256) {
      int k = e >> 7, n = e & 127;
      unsigned short v = Bm[(size_t)(kk + k) * N + colBase + n];
      lB[(n >> 4) * 512 + ((n & 15) + 16 * ((k >> 4) & 1)) * 16 + (k & 15)] = v;
    }
  };

  stageA(0);
  stageB(0);
  async_wait();
  __syncthreads();

  for (int k0 = 0; k0 < K; k0 += 32) {
    bf16x16 af[2], bfm[4];
    for (int i = 0; i < 2; i++)
      af[i] = *reinterpret_cast<const bf16x16*>(&lA[(wm * 2 + i) * 512 + lane * 16]);
    for (int j = 0; j < 4; j++)
      bfm[j] = *reinterpret_cast<const bf16x16*>(&lB[(wn * 4 + j) * 512 + lane * 16]);
    __syncthreads();                       // fragments in regs; LDS reusable
    const bool more = (k0 + 32 < K);
    if (more) {
      __builtin_prefetch(&Bm[(size_t)(k0 + 32 + (tid & 31)) * N + colBase]);
      stageA(k0 + 32);                     // async engine overlaps WMMA below
      stageB(k0 + 32);
    }
    for (int i = 0; i < 2; i++)
      for (int j = 0; j < 4; j++)
        acc[i][j] = wmma_bf16(af[i], bfm[j], acc[i][j]);
    if (more) async_wait();
    __syncthreads();
  }

  for (int i = 0; i < 2; i++)
    for (int j = 0; j < 4; j++) {
      int r0 = rowBase + wm * 32 + i * 16 + 8 * (lane >> 4);
      int c0 = colBase + wn * 64 + j * 16 + (lane & 15);
      float bcol = bias[c0];
      for (int r = 0; r < 8; r++) {
        size_t idx = (size_t)(r0 + r) * N + c0;
        float v = acc[i][j][r] + bcol;
        if (GELU) v = 0.5f * v * (1.0f + erff(v * 0.70710678118654752f));
        if (RES) v += res[idx];
        if (OUTBF) outB[idx] = f2bf(v);
        else       outF[idx] = v;
      }
    }
}

// ---------------------------------------------------------------------------
// RoPE (+ head split/transpose + optional scale): f32 [B*S,D] -> bf16 [B,H,S,HD]
// Score scale 1/sqrt(HD)=0.125 is folded into Q here (exact in bf16).
// ---------------------------------------------------------------------------
template <int ROPE>
__global__ __launch_bounds__(1024) void rope_kernel(const float* __restrict__ src,
                                                    unsigned short* __restrict__ dst,
                                                    float scale) {
  const int c = threadIdx.x;                 // h*64+hd
  const int bs = blockIdx.x;                 // 0..B*S-1
  const int b = bs / S_, s = bs % S_;
  const int h = c >> 6, hd = c & 63;
  float v = src[(size_t)bs * D_ + c];
  float o = v;
  if (ROPE && hd < 32) {
    float partner = src[(size_t)bs * D_ + (c ^ 1)];
    int fi = hd >> 1;
    float ang = (float)s * __powf(10000.0f, -(float)(2 * fi) * (1.0f / 32.0f));
    float sn, cs; __sincosf(ang, &sn, &cs);
    float rot = (hd & 1) ? partner : -partner;
    o = v * cs + rot * sn;
  }
  dst[(((size_t)(b * H_ + h)) * S_ + s) * HD_ + hd] = f2bf(o * scale);
}

// ---------------------------------------------------------------------------
// Flash attention: (b,h,64-query-rows) per 128-thread block (4 waves).
// Q,K staged by async b128 in WMMA fragment order; V staged by TDM
// tensor_load_to_lds (contiguous 8KB tile, 1-D descriptor) with async
// fallback; online softmax with wave32 shuffle reductions.
// ---------------------------------------------------------------------------
__global__ __launch_bounds__(128) void attn_kernel(
    const unsigned short* __restrict__ qg, const unsigned short* __restrict__ kg,
    const unsigned short* __restrict__ vg, const int* __restrict__ mask,
    unsigned short* __restrict__ out) {
  __shared__ __attribute__((aligned(16))) unsigned short lQ[8 * 512];
  __shared__ __attribute__((aligned(16))) unsigned short lK[8 * 512];
  __shared__ __attribute__((aligned(16))) unsigned short lVrow[64 * 64]; // row-major
  __shared__ __attribute__((aligned(16))) unsigned short lP[4 * 1024];
  __shared__ float lM[64];

  const int tid = threadIdx.x, lane = tid & 31, wave = tid >> 5;
  const int qblk = blockIdx.x & 31;
  const int bh = blockIdx.x >> 5;
  const int b = bh >> 4, h = bh & 15;
  const size_t head_off = ((size_t)(b * H_ + h)) * S_ * HD_;

  // ---- Q: async, fragment order (A-type). 512 chunks / 128 thr = 4 each.
  for (int e = tid; e < 512; e += 128) {
    int f = e >> 6, ls = (e >> 1) & 31, c = e & 1;
    int subtile = f >> 1, p = f & 1;
    int row = subtile * 16 + (ls & 15);
    int hd = p * 32 + 16 * (ls >> 4) + 8 * c;
    async_b128(lds_addr_of(&lQ[f * 512 + ls * 16 + c * 8]),
               &qg[head_off + (size_t)(qblk * 64 + row) * HD_ + hd]);
  }
  async_wait();
  __syncthreads();
  bf16x16 qf0 = *reinterpret_cast<const bf16x16*>(&lQ[(wave * 2 + 0) * 512 + lane * 16]);
  bf16x16 qf1 = *reinterpret_cast<const bf16x16*>(&lQ[(wave * 2 + 1) * 512 + lane * 16]);

  floatx8 O[4];
  for (int t = 0; t < 4; t++) O[t] = {};
  float mrow[8], lrow[8];
  for (int r = 0; r < 8; r++) { mrow[r] = -3.0e38f; lrow[r] = 0.f; }
  __syncthreads();

  for (int kb = 0; kb < S_ / 64; kb++) {
    // ---- K: async, fragment order (B-type for QK^T: kdim=hd, n=key)
    for (int e = tid; e < 512; e += 128) {
      int f = e >> 6, ls = (e >> 1) & 31, c = e & 1;
      int t = f >> 1, p = f & 1;
      int key = t * 16 + (ls & 15);
      int hd = p * 32 + 16 * (ls >> 4) + 8 * c;
      async_b128(lds_addr_of(&lK[f * 512 + ls * 16 + c * 8]),
                 &kg[head_off + (size_t)(kb * 64 + key) * HD_ + hd]);
    }
    // ---- V: contiguous 8KB tile -> TDM (wave 0 issues; EXEC-ignored op)
#if __has_builtin(__builtin_amdgcn_tensor_load_to_lds)
    if (wave == 0) {
      unsigned long long ga =
          (unsigned long long)(const void*)(vg + head_off + (size_t)(kb * 64) * HD_);
      uint32x4 g0 = { 1u,                                   // count=1, user mode
                      lds_addr_of(&lVrow[0]),               // lds_addr
                      (unsigned)(ga & 0xFFFFFFFFu),         // global_addr lo
                      (unsigned)((ga >> 32) & 0x01FFFFFFu) | (2u << 30) }; // type=2
      int32x8 g1 = { (int)(1u << 16),       // data_size = 2B
                     (int)(4096u << 16),    // tensor_dim0 = 4096 (1-D view)
                     (int)(1u << 16),       // tensor_dim1 = 1
                     (int)(4096u << 16),    // tile_dim0 = 4096
                     1,                     // tile_dim1 = 1
                     4096, 0, 0 };          // tensor_dim0_stride = 4096
      int32x4 z4 = { 0, 0, 0, 0 };
#if __clang_major__ >= 23
      int32x8 z8 = { 0, 0, 0, 0, 0, 0, 0, 0 };
      __builtin_amdgcn_tensor_load_to_lds(g0, g1, z4, z4, z8, 0);
#else
      __builtin_amdgcn_tensor_load_to_lds(g0, g1, z4, z4, 0);
#endif
#if __has_builtin(__builtin_amdgcn_s_wait_tensorcnt)
      __builtin_amdgcn_s_wait_tensorcnt((short)0);
#else
      asm volatile("s_wait_tensorcnt 0x0" ::: "memory");
#endif
    }
#else
    for (int e = tid; e < 256; e += 128)    // async fallback, contiguous
      async_b128(lds_addr_of(&lVrow[e * 8]),
                 &vg[head_off + (size_t)(kb * 64) * HD_ + e * 8]);
#endif
    if (tid < 64) lM[tid] = (mask[b * S_ + kb * 64 + tid] == 0) ? -1.0e9f : 0.0f;
    async_wait();
    __syncthreads();

    // ---- scores
    floatx8 Sf[4];
    for (int t = 0; t < 4; t++) {
      bf16x16 b0 = *reinterpret_cast<const bf16x16*>(&lK[(t * 2 + 0) * 512 + lane * 16]);
      bf16x16 b1 = *reinterpret_cast<const bf16x16*>(&lK[(t * 2 + 1) * 512 + lane * 16]);
      floatx8 c = {};
      c = wmma_bf16(qf0, b0, c);
      c = wmma_bf16(qf1, b1, c);
      float madd = lM[t * 16 + (lane & 15)];
      for (int r = 0; r < 8; r++) c[r] += madd;
      Sf[t] = c;
    }

    // ---- online softmax (row = r + 8*(lane>>4))
    for (int r = 0; r < 8; r++) {
      float rm = Sf[0][r];
      for (int t = 1; t < 4; t++) rm = fmaxf(rm, Sf[t][r]);
      for (int m = 8; m >= 1; m >>= 1) rm = fmaxf(rm, __shfl_xor(rm, m, 32));
      float mn = fmaxf(mrow[r], rm);
      float sc = __expf(mrow[r] - mn);
      mrow[r] = mn;
      float ps = 0.f;
      for (int t = 0; t < 4; t++) {
        float p = __expf(Sf[t][r] - mn);
        Sf[t][r] = p;
        ps += p;
      }
      for (int m = 8; m >= 1; m >>= 1) ps += __shfl_xor(ps, m, 32);
      lrow[r] = lrow[r] * sc + ps;
      for (int t = 0; t < 4; t++) O[t][r] *= sc;
    }

    // ---- repack P (C layout) -> A-fragment layout via per-wave LDS slab
    for (int t = 0; t < 4; t++)
      for (int r = 0; r < 8; r++) {
        int row = r + 8 * (lane >> 4);
        int col = t * 16 + (lane & 15);
        int kk = col & 31;
        int grp = (kk >> 4) & 1;
        int slot = kk & 15;
        lP[wave * 1024 + (col >> 5) * 512 + (row + 16 * grp) * 16 + slot] =
            f2bf(Sf[t][r]);
      }
    bf16x16 pf0 = *reinterpret_cast<const bf16x16*>(&lP[wave * 1024 + lane * 16]);
    bf16x16 pf1 = *reinterpret_cast<const bf16x16*>(&lP[wave * 1024 + 512 + lane * 16]);

    // ---- P*V: gather V B-fragments from row-major LDS slab
    for (int t = 0; t < 4; t++) {
      union { bf16x16 v; unsigned short u[16]; } vf0, vf1;
      int hd = t * 16 + (lane & 15);
      int kb16 = 16 * (lane >> 4);
#pragma unroll
      for (int j = 0; j < 16; j++) {
        vf0.u[j] = lVrow[(kb16 + j) * 64 + hd];          // keys 0..31 part
        vf1.u[j] = lVrow[(32 + kb16 + j) * 64 + hd];     // keys 32..63 part
      }
      O[t] = wmma_bf16(pf0, vf0.v, O[t]);
      O[t] = wmma_bf16(pf1, vf1.v, O[t]);
    }
    __syncthreads();
  }

  // ---- normalize, write [B,S,D] bf16 (head transpose fused)
  for (int t = 0; t < 4; t++)
    for (int r = 0; r < 8; r++) {
      int row = qblk * 64 + wave * 16 + r + 8 * (lane >> 4);
      int col = t * 16 + (lane & 15);
      float v = O[t][r] / lrow[r];
      out[((size_t)(b * S_ + row)) * D_ + h * HD_ + col] = f2bf(v);
    }
}

// ---------------------------------------------------------------------------
// Host launcher
// ---------------------------------------------------------------------------
extern "C" void kernel_launch(void* const* d_in, const int* in_sizes, int n_in,
                              void* d_out, int out_size, void* d_ws, size_t ws_size,
                              hipStream_t stream) {
  (void)in_sizes; (void)n_in; (void)out_size; (void)ws_size;
  const float* x    = (const float*)d_in[0];
  const int*   mask = (const int*)  d_in[1];
  const float* wq   = (const float*)d_in[2];
  const float* bq   = (const float*)d_in[3];
  const float* wk   = (const float*)d_in[4];
  const float* bk   = (const float*)d_in[5];
  const float* wv   = (const float*)d_in[6];
  const float* bv   = (const float*)d_in[7];
  const float* wo   = (const float*)d_in[8];
  const float* bo   = (const float*)d_in[9];
  const float* ln1w = (const float*)d_in[10];
  const float* ln1b = (const float*)d_in[11];
  const float* ln2w = (const float*)d_in[12];
  const float* ln2b = (const float*)d_in[13];
  const float* w1   = (const float*)d_in[14];
  const float* b1   = (const float*)d_in[15];
  const float* w2   = (const float*)d_in[16];
  const float* b2   = (const float*)d_in[17];
  float* out = (float*)d_out;

  char* ws = (char*)d_ws;
  auto alloc = [&](size_t bytes) -> char* {
    char* p = ws;
    ws += (bytes + 255) & ~(size_t)255;
    return p;
  };
  unsigned short* wq_b = (unsigned short*)alloc((size_t)D_ * D_ * 2);
  unsigned short* wk_b = (unsigned short*)alloc((size_t)D_ * D_ * 2);
  unsigned short* wv_b = (unsigned short*)alloc((size_t)D_ * D_ * 2);
  unsigned short* wo_b = (unsigned short*)alloc((size_t)D_ * D_ * 2);
  unsigned short* w1_b = (unsigned short*)alloc((size_t)D_ * FF_ * 2);
  unsigned short* w2_b = (unsigned short*)alloc((size_t)FF_ * D_ * 2);
  unsigned short* h_b  = (unsigned short*)alloc((size_t)MROWS * D_ * 2);
  float* qf = (float*)alloc((size_t)MROWS * D_ * 4);
  float* kf = (float*)alloc((size_t)MROWS * D_ * 4);
  float* vf = (float*)alloc((size_t)MROWS * D_ * 4);
  unsigned short* qb = (unsigned short*)alloc((size_t)MROWS * D_ * 2);
  unsigned short* kb = (unsigned short*)alloc((size_t)MROWS * D_ * 2);
  unsigned short* vb = (unsigned short*)alloc((size_t)MROWS * D_ * 2);
  unsigned short* ao = (unsigned short*)alloc((size_t)MROWS * D_ * 2);
  float* x2 = (float*)alloc((size_t)MROWS * D_ * 4);
  unsigned short* h2_b = (unsigned short*)alloc((size_t)MROWS * D_ * 2);
  unsigned short* g_b  = (unsigned short*)alloc((size_t)MROWS * FF_ * 2);

  const int nDD = D_ * D_, nDF = D_ * FF_;
  cvt_kernel<<<(nDD + 255) / 256, 256, 0, stream>>>(wq, wq_b, nDD);
  cvt_kernel<<<(nDD + 255) / 256, 256, 0, stream>>>(wk, wk_b, nDD);
  cvt_kernel<<<(nDD + 255) / 256, 256, 0, stream>>>(wv, wv_b, nDD);
  cvt_kernel<<<(nDD + 255) / 256, 256, 0, stream>>>(wo, wo_b, nDD);
  cvt_kernel<<<(nDF + 255) / 256, 256, 0, stream>>>(w1, w1_b, nDF);
  cvt_kernel<<<(nDF + 255) / 256, 256, 0, stream>>>(w2, w2_b, nDF);

  ln_kernel<<<MROWS, 256, 0, stream>>>(x, ln1w, ln1b, h_b);

  dim3 gDD(D_ / 128, MROWS / 128);
  gemm_kernel<0, 0, 0><<<gDD, 256, 0, stream>>>(h_b, wq_b, bq, nullptr, qf, nullptr,
                                                MROWS, D_, D_);
  gemm_kernel<0, 0, 0><<<gDD, 256, 0, stream>>>(h_b, wk_b, bk, nullptr, kf, nullptr,
                                                MROWS, D_, D_);
  gemm_kernel<0, 0, 0><<<gDD, 256, 0, stream>>>(h_b, wv_b, bv, nullptr, vf, nullptr,
                                                MROWS, D_, D_);

  rope_kernel<1><<<MROWS, 1024, 0, stream>>>(qf, qb, 0.125f);  // fold 1/sqrt(HD)
  rope_kernel<1><<<MROWS, 1024, 0, stream>>>(kf, kb, 1.0f);
  rope_kernel<0><<<MROWS, 1024, 0, stream>>>(vf, vb, 1.0f);

  attn_kernel<<<B_ * H_ * (S_ / 64), 128, 0, stream>>>(qb, kb, vb, mask, ao);

  gemm_kernel<0, 1, 0><<<gDD, 256, 0, stream>>>(ao, wo_b, bo, x, x2, nullptr,
                                                MROWS, D_, D_);

  ln_kernel<<<MROWS, 256, 0, stream>>>(x2, ln2w, ln2b, h2_b);

  dim3 gDF(FF_ / 128, MROWS / 128);
  gemm_kernel<1, 0, 1><<<gDF, 256, 0, stream>>>(h2_b, w1_b, b1, nullptr, nullptr, g_b,
                                                MROWS, FF_, D_);

  gemm_kernel<0, 1, 0><<<gDD, 256, 0, stream>>>(g_b, w2_b, b2, x2, out, nullptr,
                                                MROWS, D_, FF_);
}
